// AVSegHead_40673340293645
// MI455X (gfx1250) — matrix operand order repaired and checked by reading, not compiled, not run on hardware
//
#include <hip/hip_runtime.h>
#include <hip/hip_bf16.h>
#include <math.h>

// ---------------------------------------------------------------------------
// CDNA5 (gfx1250) implementation of the AVSegHead reference.
// Dense contractions: v_wmma_f32_16x16x32_f16 with 32x32 per-wave tiles
// (2x2 fragment reuse -> 4 WMMAs per K-step, 2x arithmetic intensity).
// Softmax / layernorm / masking stay in fp32 VALU with LDS reductions.
// ---------------------------------------------------------------------------

typedef __attribute__((ext_vector_type(16))) _Float16 v16h;
typedef __attribute__((ext_vector_type(8)))  float    v8f;
typedef __attribute__((ext_vector_type(4)))  float    vf4;
using h2 = decltype(__builtin_amdgcn_cvt_pkrtz(0.0f, 0.0f));

#define DEVFN static __device__ __forceinline__

constexpr int BB  = 2;
constexpr int QQ  = 300;
constexpr int EE  = 256;
constexpr int LLY = 3;
constexpr int NHH = 8;
constexpr int DHD = 32;
constexpr int FFF = 2048;
constexpr int HW0 = 64 * 64;   // 4096
constexpr int HW1 = 32 * 32;   // 1024
constexpr int HW2 = 16 * 16;   // 256
constexpr int TTK = HW0 + HW1 + HW2;  // 5376
constexpr float RAD = 0.18f;

DEVFN int imin(int a, int b) { return a < b ? a : b; }
DEVFN int imax(int a, int b) { return a > b ? a : b; }
DEVFN h2  pkh(float a, float b) { return __builtin_amdgcn_cvt_pkrtz(a, b); }

union FragU { v16h v; h2 h[8]; };

// 16 fp32 values (as 4x float4, packed in order) -> f16 fragment via v_cvt_pk
DEVFN v16h cvt16(vf4 a, vf4 b, vf4 c, vf4 d) {
  FragU u;
  u.h[0] = pkh(a[0], a[1]); u.h[1] = pkh(a[2], a[3]);
  u.h[2] = pkh(b[0], b[1]); u.h[3] = pkh(b[2], b[3]);
  u.h[4] = pkh(c[0], c[1]); u.h[5] = pkh(c[2], c[3]);
  u.h[6] = pkh(d[0], d[1]); u.h[7] = pkh(d[2], d[3]);
  return u.v;
}

// A fragment from a unit-stride row: two 8-float runs at +kg*8 and +16+kg*8.
DEVFN v16h ldA(const float* row, int k0, int kg) {
  const float* p = row + k0 + kg * 8;
  return cvt16(*(const vf4*)(p), *(const vf4*)(p + 4),
               *(const vf4*)(p + 16), *(const vf4*)(p + 20));
}
// B fragment from a unit-stride column: two 8-float runs at +kg*8 and +16+kg*8.
DEVFN v16h ldB(const float* col, int k0, int kg) {
  const float* p = col + k0 + kg * 8;
  return cvt16(*(const vf4*)(p), *(const vf4*)(p + 4),
               *(const vf4*)(p + 16), *(const vf4*)(p + 20));
}

// Generic clamp-gather K-step. Loads are UNCONDITIONAL (clamped in-bounds
// index); masking is a multiply so LLVM cannot sink the load under a branch.
DEVFN v8f gstep(const float* __restrict__ Ab, const float* __restrict__ Bb,
                int row_m, int col_n, int M, int N, int K, int k0,
                long long a_sm, long long a_sk, long long b_sn, long long b_sk,
                int kg, v8f acc) {
  float fa[16], fb[16];
  const long long abase = (long long)imin(row_m, M - 1) * a_sm;
  const long long bbase = (long long)imin(col_n, N - 1) * b_sn;
  const float am = (row_m < M) ? 1.0f : 0.0f;
  const float bm = (col_n < N) ? 1.0f : 0.0f;
#pragma unroll
  for (int j = 0; j < 16; ++j) {
    // A 16x32 f16 layout: j<8 -> K = kg*8+j ; j>=8 -> K = 16+kg*8+(j-8)
    const int ka = (j < 8) ? (kg * 8 + j) : (16 + kg * 8 + (j - 8));
    const int kA = k0 + ka;
    const float av = Ab[abase + (long long)imin(kA, K - 1) * a_sk];
    fa[j] = av * ((kA < K) ? am : 0.0f);
    // B 32x16 f16 layout: same K order as A
    const int kB = k0 + ka;
    const float bv = Bb[bbase + (long long)imin(kB, K - 1) * b_sk];
    fb[j] = bv * ((kB < K) ? bm : 0.0f);
  }
  FragU ua, ub;
#pragma unroll
  for (int j = 0; j < 8; ++j) {
    ua.h[j] = pkh(fa[2 * j], fa[2 * j + 1]);
    ub.h[j] = pkh(fb[2 * j], fb[2 * j + 1]);
  }
  return __builtin_amdgcn_wmma_f32_16x16x32_f16(false, ua.v, false, ub.v,
                                                (short)0, acc, false, false);
}

DEVFN void store_tile(float* __restrict__ Cb, v8f acc, int mbase, int n,
                      int M, int N, long long c_sm, long long c_sn,
                      const float* __restrict__ bias, int relu, int kg) {
  if (n < N) {
    const float bi = bias ? bias[n] : 0.0f;
#pragma unroll
    for (int p = 0; p < 8; ++p) {     // f32 C/D: VGPR p -> M=p (+8 hi lanes)
      const int m = mbase + p + 8 * kg;
      if (m < M) {
        float v = acc[p] + bi;
        if (relu) v = fmaxf(v, 0.0f);
        Cb[(long long)m * c_sm + (long long)n * c_sn] = v;
      }
    }
  }
}

// ---------------------------------------------------------------------------
// Generic strided/batched WMMA GEMM: C[b,z,m,n] = act(sum_k A*B + bias[n])
// One 32x32 super-tile per wave (2x2 16x16 sub-tiles, fragments reused),
// 8 waves/block. Fast path: unit-stride K + full super-tile -> 16x b128
// loads feed 4 WMMAs per K-step, no predication.
// ---------------------------------------------------------------------------
__global__ __launch_bounds__(256) void wmma_gemm_f32(
    const float* __restrict__ A, const float* __restrict__ Bw,
    const float* __restrict__ bias, float* __restrict__ C,
    int M, int N, int K,
    long long a_sm, long long a_sk, long long a_sb, long long a_sz,
    long long b_sn, long long b_sk, long long b_sb, long long b_sz,
    long long c_sm, long long c_sn, long long c_sb, long long c_sz,
    int relu)
{
  const int wave = threadIdx.x >> 5;
  const int lane = threadIdx.x & 31;
  const int st_n = (N + 31) >> 5;
  const long long nst = (long long)((M + 31) >> 5) * st_n;
  const long long stile = (long long)blockIdx.x * 8 + wave;
  if (stile >= nst) return;              // wave-uniform
  const int b = blockIdx.y, z = blockIdx.z;
  const int sm = (int)(stile / st_n), sn = (int)(stile % st_n);
  const float* Ab = A + (long long)b * a_sb + (long long)z * a_sz;
  const float* Bb = Bw + (long long)b * b_sb + (long long)z * b_sz;

  const int kg  = lane >> 4;
  const int l15 = lane & 15;
  const int m0 = sm * 32 + l15, m1 = m0 + 16;
  const int n0 = sn * 32 + l15, n1 = n0 + 16;

  v8f a00 = {}, a01 = {}, a10 = {}, a11 = {};
  const bool fast = (a_sk == 1) && (b_sk == 1) &&
                    (sm * 32 + 32 <= M) && (sn * 32 + 32 <= N);
  if (fast) {
    const float* ar0 = Ab + (long long)m0 * a_sm;
    const float* ar1 = Ab + (long long)m1 * a_sm;
    const float* br0 = Bb + (long long)n0 * b_sn;
    const float* br1 = Bb + (long long)n1 * b_sn;
    int k0 = 0;
    for (; k0 + 32 <= K; k0 += 32) {
      if (k0 + 64 <= K) {                      // global_prefetch_b8
        __builtin_prefetch(ar0 + k0 + 32, 0, 1);
        __builtin_prefetch(ar1 + k0 + 32, 0, 1);
        __builtin_prefetch(br0 + k0 + 32, 0, 1);
        __builtin_prefetch(br1 + k0 + 32, 0, 1);
      }
      const v16h af0 = ldA(ar0, k0, kg);
      const v16h af1 = ldA(ar1, k0, kg);
      const v16h bf0 = ldB(br0, k0, kg);
      const v16h bf1 = ldB(br1, k0, kg);
      a00 = __builtin_amdgcn_wmma_f32_16x16x32_f16(false, af0, false, bf0, (short)0, a00, false, false);
      a01 = __builtin_amdgcn_wmma_f32_16x16x32_f16(false, af0, false, bf1, (short)0, a01, false, false);
      a10 = __builtin_amdgcn_wmma_f32_16x16x32_f16(false, af1, false, bf0, (short)0, a10, false, false);
      a11 = __builtin_amdgcn_wmma_f32_16x16x32_f16(false, af1, false, bf1, (short)0, a11, false, false);
    }
    if (k0 < K) {
      a00 = gstep(Ab, Bb, m0, n0, M, N, K, k0, a_sm, 1, b_sn, 1, kg, a00);
      a01 = gstep(Ab, Bb, m0, n1, M, N, K, k0, a_sm, 1, b_sn, 1, kg, a01);
      a10 = gstep(Ab, Bb, m1, n0, M, N, K, k0, a_sm, 1, b_sn, 1, kg, a10);
      a11 = gstep(Ab, Bb, m1, n1, M, N, K, k0, a_sm, 1, b_sn, 1, kg, a11);
    }
  } else {
    for (int k0 = 0; k0 < K; k0 += 32) {
      a00 = gstep(Ab, Bb, m0, n0, M, N, K, k0, a_sm, a_sk, b_sn, b_sk, kg, a00);
      a01 = gstep(Ab, Bb, m0, n1, M, N, K, k0, a_sm, a_sk, b_sn, b_sk, kg, a01);
      a10 = gstep(Ab, Bb, m1, n0, M, N, K, k0, a_sm, a_sk, b_sn, b_sk, kg, a10);
      a11 = gstep(Ab, Bb, m1, n1, M, N, K, k0, a_sm, a_sk, b_sn, b_sk, kg, a11);
    }
  }

  float* Cb = C + (long long)b * c_sb + (long long)z * c_sz;
  store_tile(Cb, a00, sm * 32,      n0, M, N, c_sm, c_sn, bias, relu, kg);
  store_tile(Cb, a01, sm * 32,      n1, M, N, c_sm, c_sn, bias, relu, kg);
  store_tile(Cb, a10, sm * 32 + 16, n0, M, N, c_sm, c_sn, bias, relu, kg);
  store_tile(Cb, a11, sm * 32 + 16, n1, M, N, c_sm, c_sn, bias, relu, kg);
}

// ---------------------------------------------------------------------------
// 3x3 conv (pad 1) as WMMA GEMM with inline im2col gather (branch-free:
// clamped position + multiply-mask so the loads stay unconditional).
// Hin [B,4096,256] pos-major -> Src [B,256,4096] (ch-major) and
// SrcT [B,4096,256] (pos-major, for the mask einsum fast path).
// K = 2304 ordered k = c*9 + ky*3 + kx to match OIHW weights.
// ---------------------------------------------------------------------------
__global__ __launch_bounds__(256) void wmma_conv3x3(
    const float* __restrict__ Hin, const float* __restrict__ W,
    const float* __restrict__ bias, float* __restrict__ Src,
    float* __restrict__ SrcT)
{
  const int wave = threadIdx.x >> 5;
  const int lane = threadIdx.x & 31;
  const int tile = blockIdx.x * 8 + wave;
  if (tile >= 256 * 16) return;
  const int b = blockIdx.y;
  const int tm = tile >> 4, tn = tile & 15;
  const int kg = lane >> 4;
  const int row_m = tm * 16 + (lane & 15);   // output position
  const int col_n = tn * 16 + (lane & 15);   // output channel
  const int py = row_m >> 6, px = row_m & 63;
  const float* Hb = Hin + (long long)b * HW0 * EE;
  const float* wrow = W + (long long)col_n * 2304;

  v8f acc = {};
  for (int k0 = 0; k0 < 2304; k0 += 32) {
    float fa[16];
#pragma unroll
    for (int j = 0; j < 16; ++j) {
      const int ka = (j < 8) ? (kg * 8 + j) : (16 + kg * 8 + (j - 8));
      const int kk = k0 + ka;
      const int c  = kk / 9, r = kk - 9 * c;
      const int ky = r / 3,  kx = r - 3 * ky;
      const int yy = py + ky - 1, xx = px + kx - 1;
      const bool inb = ((unsigned)yy < 64u) && ((unsigned)xx < 64u);
      const int pos = inb ? (yy * 64 + xx) : 0;
      const float av = Hb[(long long)pos * EE + c];   // unconditional load
      fa[j] = av * (inb ? 1.0f : 0.0f);               // mask by multiply
    }
    FragU ua;
#pragma unroll
    for (int j = 0; j < 8; ++j) ua.h[j] = pkh(fa[2 * j], fa[2 * j + 1]);
    const v16h bf = ldB(wrow, k0, kg);
    acc = __builtin_amdgcn_wmma_f32_16x16x32_f16(false, ua.v, false, bf,
                                                 (short)0, acc, false, false);
  }
  const float bi = bias[col_n];
  float* Sb = Src  + (long long)b * EE * HW0;
  float* Tb = SrcT + (long long)b * HW0 * EE;
#pragma unroll
  for (int p = 0; p < 8; ++p) {
    const int m_out = tm * 16 + p + 8 * kg;
    const float v = acc[p] + bi;
    Sb[(long long)col_n * HW0 + m_out] = v;   // channel-major
    Tb[(long long)m_out * EE + col_n]  = v;   // position-major
  }
}

// in [B,R,C] -> out [B,C,R]
__global__ void transpose_kernel(const float* __restrict__ in,
                                 float* __restrict__ out, int R, int Ccol) {
  const long long total = (long long)BB * R * Ccol;
  for (long long i = (long long)blockIdx.x * blockDim.x + threadIdx.x;
       i < total; i += (long long)gridDim.x * blockDim.x) {
    const int c = (int)(i % Ccol);
    const long long u = i / Ccol;
    const int r = (int)(u % R);
    const int b = (int)(u / R);
    out[((long long)b * Ccol + c) * R + r] = in[i];
  }
}

// vin [B,T,E] -> vout [B,NH,DH,T]  (unit-stride K for the attn*V GEMM)
__global__ void vtrans_kernel(const float* __restrict__ vin,
                              float* __restrict__ vout, int T) {
  const long long total = (long long)BB * T * EE;
  for (long long i = (long long)blockIdx.x * blockDim.x + threadIdx.x;
       i < total; i += (long long)gridDim.x * blockDim.x) {
    const int e = (int)(i % EE);
    const long long u = i / EE;
    const int t = (int)(u % T);
    const int b = (int)(u / T);
    const int h = e >> 5, d = e & 31;
    vout[(((long long)b * NHH + h) * DHD + d) * T + t] = vin[i];
  }
}

DEVFN float bilin_sample(const float* mem, int b, int tbase, int S, int e,
                         float fy, float fx) {
  int x0 = (int)floorf(fx), y0 = (int)floorf(fy);
  const float wx = fx - (float)x0, wy = fy - (float)y0;
  int x1 = imin(imax(x0 + 1, 0), S - 1), y1 = imin(imax(y0 + 1, 0), S - 1);
  x0 = imin(imax(x0, 0), S - 1);         y0 = imin(imax(y0, 0), S - 1);
  const float* base = mem + ((long long)b * TTK + tbase) * EE + e;
  const float v00 = base[(long long)(y0 * S + x0) * EE];
  const float v01 = base[(long long)(y0 * S + x1) * EE];
  const float v10 = base[(long long)(y1 * S + x0) * EE];
  const float v11 = base[(long long)(y1 * S + x1) * EE];
  return v00 * (1.f - wx) * (1.f - wy) + v01 * wx * (1.f - wy) +
         v10 * (1.f - wx) * wy         + v11 * wx * wy;
}

__global__ void build_cat_kernel(const float* __restrict__ mem,
                                 float* __restrict__ cat) {
  const long long total = (long long)BB * HW0 * 768;
  for (long long i = (long long)blockIdx.x * blockDim.x + threadIdx.x;
       i < total; i += (long long)gridDim.x * blockDim.x) {
    const int k   = (int)(i % 768);
    const long long u = i / 768;
    const int pos = (int)(u % HW0);
    const int b   = (int)(u / HW0);
    const int py = pos >> 6, px = pos & 63;
    float v;
    if (k < 256) {
      v = mem[((long long)b * TTK + pos) * EE + k];
    } else if (k < 512) {
      const float fy = (py + 0.5f) * 0.5f - 0.5f, fx = (px + 0.5f) * 0.5f - 0.5f;
      v = bilin_sample(mem, b, HW0, 32, k - 256, fy, fx);
    } else {
      const float fy = (py + 0.5f) * 0.25f - 0.5f, fx = (px + 0.5f) * 0.25f - 0.5f;
      v = bilin_sample(mem, b, HW0 + HW1, 16, k - 512, fy, fx);
    }
    cat[i] = v;
  }
}

DEVFN void token_coord(int t, float& tx, float& ty) {
  if (t < HW0)      { const int y = t >> 6, x = t & 63;
                      tx = (x + 0.5f) / 64.f; ty = (y + 0.5f) / 64.f; }
  else if (t < HW0 + HW1) { const int u = t - HW0; const int y = u >> 5, x = u & 31;
                      tx = (x + 0.5f) / 32.f; ty = (y + 0.5f) / 32.f; }
  else              { const int u = t - HW0 - HW1; const int y = u >> 4, x = u & 15;
                      tx = (x + 0.5f) / 16.f; ty = (y + 0.5f) / 16.f; }
}

__global__ __launch_bounds__(256) void extra_prep_kernel(
    const float* __restrict__ centroids, const unsigned char* __restrict__ valid,
    int* __restrict__ amin, int* __restrict__ anyvis) {
  __shared__ float sd[256]; __shared__ int si[256]; __shared__ int sa[256];
  const int bq = blockIdx.x;
  const float cx = centroids[(long long)bq * 2 + 0];
  const float cy = centroids[(long long)bq * 2 + 1];
  const int val = valid[bq] ? 1 : 0;
  float best = 3.4e38f; int bidx = 0; int any = 0;
  for (int t = threadIdx.x; t < TTK; t += 256) {
    float tx, ty; token_coord(t, tx, ty);
    const float dx = cx - tx, dy = cy - ty;
    const float d2 = dx * dx + dy * dy;
    if (d2 < best) { best = d2; bidx = t; }
    if (val && fabsf(dx) <= RAD && fabsf(dy) <= RAD) any = 1;
  }
  sd[threadIdx.x] = best; si[threadIdx.x] = bidx; sa[threadIdx.x] = any;
  __syncthreads();
  for (int s = 128; s > 0; s >>= 1) {
    if (threadIdx.x < s) {
      if (sd[threadIdx.x + s] < sd[threadIdx.x]) {
        sd[threadIdx.x] = sd[threadIdx.x + s];
        si[threadIdx.x] = si[threadIdx.x + s];
      }
      sa[threadIdx.x] |= sa[threadIdx.x + s];
    }
    __syncthreads();
  }
  if (threadIdx.x == 0) { amin[bq] = si[0]; anyvis[bq] = sa[0]; }
}

__global__ void build_extra_kernel(
    const float* __restrict__ centroids, const unsigned char* __restrict__ valid,
    const int* __restrict__ amin, const int* __restrict__ anyvis,
    float* __restrict__ extra) {
  const long long total = (long long)BB * QQ * TTK;
  for (long long i = (long long)blockIdx.x * blockDim.x + threadIdx.x;
       i < total; i += (long long)gridDim.x * blockDim.x) {
    const int t = (int)(i % TTK);
    const int bq = (int)(i / TTK);
    const float cx = centroids[(long long)bq * 2 + 0];
    const float cy = centroids[(long long)bq * 2 + 1];
    float tx, ty; token_coord(t, tx, ty);
    const float dx = cx - tx, dy = cy - ty;
    const int val = valid[bq] ? 1 : 0;
    bool vis = (fabsf(dx) <= RAD) && (fabsf(dy) <= RAD) && val;
    if (!anyvis[bq]) vis = (t == amin[bq]);
    if (!val) vis = true;
    const float mask = vis ? 0.0f : -10000.0f;
    const float sx = dx / RAD, sy = dy / RAD;
    const float bia = val ? (-2.0f * (sx * sx + sy * sy)) : 0.0f;
    extra[i] = mask + bia;
  }
}

__global__ __launch_bounds__(256) void pool_kernel(
    const float* __restrict__ src, const float* __restrict__ centroids,
    const unsigned char* __restrict__ valid, float* __restrict__ pooled) {
  const int bq = blockIdx.x;
  const int b = bq / QQ;
  const int c = threadIdx.x;
  if (!valid[bq]) { pooled[(long long)bq * EE + c] = 0.0f; return; }
  const float cx = centroids[(long long)bq * 2 + 0];
  const float cy = centroids[(long long)bq * 2 + 1];
  const float xc = fminf(fmaxf(cx * 64.f - 0.5f, 0.f), 63.f);
  const float yc = fminf(fmaxf(cy * 64.f - 0.5f, 0.f), 63.f);
  const int xlo = imax(0, (int)ceilf(xc - 12.f)), xhi = imin(63, (int)floorf(xc + 12.f));
  const int ylo = imax(0, (int)ceilf(yc - 12.f)), yhi = imin(63, (int)floorf(yc + 12.f));
  const float* s = src + ((long long)b * EE + c) * HW0;
  float sum = 0.0f;
  for (int y = ylo; y <= yhi; ++y)
    for (int x = xlo; x <= xhi; ++x)
      sum += s[y * 64 + x];
  const float cnt = (float)((xhi - xlo + 1) * (yhi - ylo + 1));
  pooled[(long long)bq * EE + c] = sum / fmaxf(cnt, 1.0f);
}

__global__ void build_qc_kernel(const float* __restrict__ qe,
                                const float* __restrict__ omics,
                                const float* __restrict__ pooled,
                                float* __restrict__ qc) {
  const long long total = (long long)BB * QQ * 768;
  for (long long i = (long long)blockIdx.x * blockDim.x + threadIdx.x;
       i < total; i += (long long)gridDim.x * blockDim.x) {
    const int k = (int)(i % 768);
    const long long bq = i / 768;
    float v;
    if (k < 256)      v = qe[bq * EE + k];
    else if (k < 512) v = omics[bq * EE + (k - 256)];
    else              v = pooled[bq * EE + (k - 512)];
    qc[i] = v;
  }
}

__global__ __launch_bounds__(256) void softmax_kernel(
    float* __restrict__ S, const float* __restrict__ extra, int T, float scale) {
  __shared__ float red[256];
  const int r = blockIdx.x;               // ((b*NH)+h)*Q + q
  const int q = r % QQ;
  const int b = r / (NHH * QQ);
  float* row = S + (long long)r * T;
  const float* ex = extra ? (extra + ((long long)b * QQ + q) * T) : nullptr;
  float m = -3.4e38f;
  for (int t = threadIdx.x; t < T; t += 256)
    m = fmaxf(m, row[t] * scale + (ex ? ex[t] : 0.0f));
  red[threadIdx.x] = m; __syncthreads();
  for (int s = 128; s > 0; s >>= 1) {
    if (threadIdx.x < s) red[threadIdx.x] = fmaxf(red[threadIdx.x], red[threadIdx.x + s]);
    __syncthreads();
  }
  m = red[0]; __syncthreads();
  float sum = 0.0f;
  for (int t = threadIdx.x; t < T; t += 256) {
    const float v = __expf(row[t] * scale + (ex ? ex[t] : 0.0f) - m);
    row[t] = v; sum += v;
  }
  red[threadIdx.x] = sum; __syncthreads();
  for (int s = 128; s > 0; s >>= 1) {
    if (threadIdx.x < s) red[threadIdx.x] += red[threadIdx.x + s];
    __syncthreads();
  }
  const float inv = 1.0f / red[0];
  for (int t = threadIdx.x; t < T; t += 256) row[t] *= inv;
}

__global__ __launch_bounds__(256) void ln_kernel(
    const float* __restrict__ x, const float* __restrict__ res,
    const float* __restrict__ g, const float* __restrict__ bt,
    float* __restrict__ out) {
  __shared__ float red[256];
  const long long row = blockIdx.x;
  const int c = threadIdx.x;
  float v = x[row * EE + c] + (res ? res[row * EE + c] : 0.0f);
  red[c] = v; __syncthreads();
  for (int s = 128; s > 0; s >>= 1) {
    if (c < s) red[c] += red[c + s];
    __syncthreads();
  }
  const float mean = red[0] * (1.0f / EE); __syncthreads();
  const float d = v - mean;
  red[c] = d * d; __syncthreads();
  for (int s = 128; s > 0; s >>= 1) {
    if (c < s) red[c] += red[c + s];
    __syncthreads();
  }
  const float var = red[0] * (1.0f / EE);
  out[row * EE + c] = d * rsqrtf(var + 1e-5f) * g[c] + bt[c];
}

// ---------------------------------------------------------------------------
static inline void launch_gemm(hipStream_t st, const float* A, const float* Bw,
    const float* bias, float* C, int M, int N, int K, int nb, int nz,
    long long a_sm, long long a_sk, long long a_sb, long long a_sz,
    long long b_sn, long long b_sk, long long b_sb, long long b_sz,
    long long c_sm, long long c_sn, long long c_sb, long long c_sz, int relu) {
  const long long tiles = (long long)((M + 31) / 32) * ((N + 31) / 32);
  dim3 grid((unsigned)((tiles + 7) / 8), (unsigned)nb, (unsigned)nz);
  wmma_gemm_f32<<<grid, dim3(256), 0, st>>>(A, Bw, bias, C, M, N, K,
      a_sm, a_sk, a_sb, a_sz, b_sn, b_sk, b_sb, b_sz,
      c_sm, c_sn, c_sb, c_sz, relu);
}

extern "C" void kernel_launch(void* const* d_in, const int* in_sizes, int n_in,
                              void* d_out, int out_size, void* d_ws, size_t ws_size,
                              hipStream_t stream) {
  (void)in_sizes; (void)n_in; (void)out_size; (void)ws_size;
  const float* x0        = (const float*)d_in[0];
  const float* x1        = (const float*)d_in[1];
  const float* x2        = (const float*)d_in[2];
  const float* omics     = (const float*)d_in[3];
  const float* centroids = (const float*)d_in[4];
  const unsigned char* valid = (const unsigned char*)d_in[5];
  const float* proj_w0 = (const float*)d_in[6];  const float* proj_b0 = (const float*)d_in[7];
  const float* proj_w1 = (const float*)d_in[8];  const float* proj_b1 = (const float*)d_in[9];
  const float* proj_w2 = (const float*)d_in[10]; const float* proj_b2 = (const float*)d_in[11];
  const float* fuse_w1 = (const float*)d_in[12]; const float* fuse_b1 = (const float*)d_in[13];
  const float* fuse_w2 = (const float*)d_in[14]; const float* fuse_b2 = (const float*)d_in[15];
  const float* qg_w1   = (const float*)d_in[16]; const float* qg_b1   = (const float*)d_in[17];
  const float* qg_w2   = (const float*)d_in[18]; const float* qg_b2   = (const float*)d_in[19];
  const float* ctx_w1  = (const float*)d_in[20]; const float* ctx_b1  = (const float*)d_in[21];
  const float* ctx_w2  = (const float*)d_in[22]; const float* ctx_b2  = (const float*)d_in[23];
  const float* ctx_ln_g = (const float*)d_in[24]; const float* ctx_ln_b = (const float*)d_in[25];
  const float* sa_w = (const float*)d_in[26]; const float* sa_b = (const float*)d_in[27];
  const float* ca_w = (const float*)d_in[28]; const float* ca_b = (const float*)d_in[29];
  const float* ffn_w1 = (const float*)d_in[30]; const float* ffn_b1 = (const float*)d_in[31];
  const float* ffn_w2 = (const float*)d_in[32]; const float* ffn_b2 = (const float*)d_in[33];
  const float* ln_g = (const float*)d_in[34]; const float* ln_b = (const float*)d_in[35];
  const float* cls_w = (const float*)d_in[36]; const float* cls_b = (const float*)d_in[37];
  const float* me_w1 = (const float*)d_in[38]; const float* me_b1 = (const float*)d_in[39];
  const float* me_w2 = (const float*)d_in[40]; const float* me_b2 = (const float*)d_in[41];
  const float* me_w3 = (const float*)d_in[42]; const float* me_b3 = (const float*)d_in[43];

  size_t off = 0;
  auto alloc = [&](size_t bytes) -> void* {
    off = (off + 255) & ~(size_t)255;
    void* p = (char*)d_ws + off;
    off += bytes;
    return p;
  };
  float* f_xt0    = (float*)alloc((size_t)BB * HW0 * 256 * 4);   // x0^T pos-major
  float* f_xt1    = (float*)alloc((size_t)BB * HW1 * 512 * 4);
  float* f_xt2    = (float*)alloc((size_t)BB * HW2 * 1024 * 4);
  float* f_mem    = (float*)alloc((size_t)BB * TTK * EE * 4);
  float* f_cat    = (float*)alloc((size_t)BB * HW0 * 768 * 4);
  float* f_h      = (float*)alloc((size_t)BB * HW0 * EE * 4);
  float* f_src    = (float*)alloc((size_t)BB * EE * HW0 * 4);    // ch-major
  float* f_srcT   = (float*)alloc((size_t)BB * HW0 * EE * 4);    // pos-major
  float* f_qeh    = (float*)alloc((size_t)BB * QQ * EE * 4);
  float* f_qe     = (float*)alloc((size_t)BB * QQ * EE * 4);
  float* f_pooled = (float*)alloc((size_t)BB * QQ * EE * 4);
  float* f_qc     = (float*)alloc((size_t)BB * QQ * 768 * 4);
  float* f_qch    = (float*)alloc((size_t)BB * QQ * EE * 4);
  float* f_tgt    = (float*)alloc((size_t)BB * QQ * EE * 4);
  float* f_t2     = (float*)alloc((size_t)BB * QQ * EE * 4);
  float* f_attnO  = (float*)alloc((size_t)BB * QQ * EE * 4);
  float* f_qp     = (float*)alloc((size_t)BB * QQ * EE * 4);
  float* f_kp     = (float*)alloc((size_t)BB * QQ * EE * 4);
  float* f_vp     = (float*)alloc((size_t)BB * QQ * EE * 4);
  float* f_kc     = (float*)alloc((size_t)BB * TTK * EE * 4);
  float* f_vc     = (float*)alloc((size_t)BB * TTK * EE * 4);
  float* f_vt     = (float*)alloc((size_t)BB * EE * TTK * 4);    // [B,NH,DH,T]
  float* f_scores = (float*)alloc((size_t)BB * NHH * QQ * TTK * 4);
  float* f_extra  = (float*)alloc((size_t)BB * QQ * TTK * 4);
  float* f_ffh    = (float*)alloc((size_t)BB * QQ * FFF * 4);
  float* f_m1     = (float*)alloc((size_t)BB * QQ * EE * 4);
  float* f_m2     = (float*)alloc((size_t)BB * QQ * EE * 4);
  float* f_m3     = (float*)alloc((size_t)BB * QQ * EE * 4);
  int*   i_amin   = (int*)alloc((size_t)BB * QQ * 4);
  int*   i_any    = (int*)alloc((size_t)BB * QQ * 4);

  const float invsq = 0.17677669529663687f;   // 1/sqrt(32)
  hipStream_t st = stream;

  // ---- 0) transpose feature maps to pos-major (unit-stride K for GEMMs) ----
  transpose_kernel<<<dim3(4096), dim3(256), 0, st>>>(x0, f_xt0, 256, HW0);
  transpose_kernel<<<dim3(2048), dim3(256), 0, st>>>(x1, f_xt1, 512, HW1);
  transpose_kernel<<<dim3(1024), dim3(256), 0, st>>>(x2, f_xt2, 1024, HW2);

  // ---- 1) per-scale projections -> memory [B,T,E] ----
  launch_gemm(st, f_xt0, proj_w0, proj_b0, f_mem, HW0, EE, 256, BB, 1,
      256, 1, (long long)HW0 * 256, 0,    256, 1, 0, 0,
      EE, 1, (long long)TTK * EE, 0, 0);
  launch_gemm(st, f_xt1, proj_w1, proj_b1, f_mem + (long long)HW0 * EE, HW1, EE, 512, BB, 1,
      512, 1, (long long)HW1 * 512, 0,    512, 1, 0, 0,
      EE, 1, (long long)TTK * EE, 0, 0);
  launch_gemm(st, f_xt2, proj_w2, proj_b2, f_mem + (long long)(HW0 + HW1) * EE, HW2, EE, 1024, BB, 1,
      1024, 1, (long long)HW2 * 1024, 0,  1024, 1, 0, 0,
      EE, 1, (long long)TTK * EE, 0, 0);

  // ---- 2) resize + concat; fuse1; 3x3 conv -> src (both layouts) ----
  build_cat_kernel<<<dim3(8192), dim3(256), 0, st>>>(f_mem, f_cat);
  launch_gemm(st, f_cat, fuse_w1, fuse_b1, f_h, HW0, EE, 768, BB, 1,
      768, 1, (long long)HW0 * 768, 0,    768, 1, 0, 0,
      EE, 1, (long long)HW0 * EE, 0, 1);
  wmma_conv3x3<<<dim3((256 * 16 + 7) / 8, BB), dim3(256), 0, st>>>(
      f_h, fuse_w2, fuse_b2, f_src, f_srcT);

  // ---- 3) query generator MLP ----
  launch_gemm(st, centroids, qg_w1, qg_b1, f_qeh, BB * QQ, EE, 2, 1, 1,
      2, 1, 0, 0,   2, 1, 0, 0,   EE, 1, 0, 0, 1);
  launch_gemm(st, f_qeh, qg_w2, qg_b2, f_qe, BB * QQ, EE, EE, 1, 1,
      EE, 1, 0, 0,  EE, 1, 0, 0,  EE, 1, 0, 0, 0);

  // ---- 4) attention mask + bias ----
  extra_prep_kernel<<<dim3(BB * QQ), dim3(256), 0, st>>>(centroids, valid, i_amin, i_any);
  build_extra_kernel<<<dim3(4096), dim3(256), 0, st>>>(centroids, valid, i_amin, i_any, f_extra);

  // ---- 5) local pooling ----
  pool_kernel<<<dim3(BB * QQ), dim3(256), 0, st>>>(f_src, centroids, valid, f_pooled);

  // ---- 6) query context fuse + LN ----
  build_qc_kernel<<<dim3(2048), dim3(256), 0, st>>>(f_qe, omics, f_pooled, f_qc);
  launch_gemm(st, f_qc, ctx_w1, ctx_b1, f_qch, BB * QQ, EE, 768, 1, 1,
      768, 1, 0, 0,  768, 1, 0, 0,  EE, 1, 0, 0, 1);
  launch_gemm(st, f_qch, ctx_w2, ctx_b2, f_t2, BB * QQ, EE, EE, 1, 1,
      EE, 1, 0, 0,   EE, 1, 0, 0,   EE, 1, 0, 0, 0);
  ln_kernel<<<dim3(BB * QQ), dim3(256), 0, st>>>(f_t2, nullptr, ctx_ln_g, ctx_ln_b, f_tgt);

  // ---- 7) decoder layers ----
  for (int i = 0; i < LLY; ++i) {
    const float* sw = sa_w + (long long)i * 4 * EE * EE;
    const float* sb = sa_b + (long long)i * 4 * EE;
    // self-attention
    launch_gemm(st, f_tgt, sw + 0 * EE * EE, sb + 0 * EE, f_qp, BB * QQ, EE, EE, 1, 1,
        EE, 1, 0, 0, EE, 1, 0, 0, EE, 1, 0, 0, 0);
    launch_gemm(st, f_tgt, sw + 1 * EE * EE, sb + 1 * EE, f_kp, BB * QQ, EE, EE, 1, 1,
        EE, 1, 0, 0, EE, 1, 0, 0, EE, 1, 0, 0, 0);
    launch_gemm(st, f_tgt, sw + 2 * EE * EE, sb + 2 * EE, f_vp, BB * QQ, EE, EE, 1, 1,
        EE, 1, 0, 0, EE, 1, 0, 0, EE, 1, 0, 0, 0);
    launch_gemm(st, f_qp, f_kp, nullptr, f_scores, QQ, QQ, DHD, BB, NHH,
        EE, 1, (long long)QQ * EE, DHD,
        EE, 1, (long long)QQ * EE, DHD,
        QQ, 1, (long long)NHH * QQ * QQ, (long long)QQ * QQ, 0);
    softmax_kernel<<<dim3(BB * NHH * QQ), dim3(256), 0, st>>>(f_scores, nullptr, QQ, invsq);
    vtrans_kernel<<<dim3(1024), dim3(256), 0, st>>>(f_vp, f_vt, QQ);
    launch_gemm(st, f_scores, f_vt, nullptr, f_attnO, QQ, DHD, QQ, BB, NHH,
        QQ, 1, (long long)NHH * QQ * QQ, (long long)QQ * QQ,
        QQ, 1, (long long)EE * QQ, (long long)DHD * QQ,
        EE, 1, (long long)QQ * EE, DHD, 0);
    launch_gemm(st, f_attnO, sw + 3 * EE * EE, sb + 3 * EE, f_t2, BB * QQ, EE, EE, 1, 1,
        EE, 1, 0, 0, EE, 1, 0, 0, EE, 1, 0, 0, 0);
    ln_kernel<<<dim3(BB * QQ), dim3(256), 0, st>>>(f_t2, f_tgt,
        ln_g + (long long)(i * 3 + 0) * EE, ln_b + (long long)(i * 3 + 0) * EE, f_tgt);

    // cross-attention
    const float* cw = ca_w + (long long)i * 4 * EE * EE;
    const float* cb = ca_b + (long long)i * 4 * EE;
    launch_gemm(st, f_tgt, cw + 0 * EE * EE, cb + 0 * EE, f_qp, BB * QQ, EE, EE, 1, 1,
        EE, 1, 0, 0, EE, 1, 0, 0, EE, 1, 0, 0, 0);
    launch_gemm(st, f_mem, cw + 1 * EE * EE, cb + 1 * EE, f_kc, BB * TTK, EE, EE, 1, 1,
        EE, 1, 0, 0, EE, 1, 0, 0, EE, 1, 0, 0, 0);
    launch_gemm(st, f_mem, cw + 2 * EE * EE, cb + 2 * EE, f_vc, BB * TTK, EE, EE, 1, 1,
        EE, 1, 0, 0, EE, 1, 0, 0, EE, 1, 0, 0, 0);
    launch_gemm(st, f_qp, f_kc, nullptr, f_scores, QQ, TTK, DHD, BB, NHH,
        EE, 1, (long long)QQ * EE, DHD,
        EE, 1, (long long)TTK * EE, DHD,
        TTK, 1, (long long)NHH * QQ * TTK, (long long)QQ * TTK, 0);
    softmax_kernel<<<dim3(BB * NHH * QQ), dim3(256), 0, st>>>(f_scores, f_extra, TTK, invsq);
    vtrans_kernel<<<dim3(4096), dim3(256), 0, st>>>(f_vc, f_vt, TTK);
    launch_gemm(st, f_scores, f_vt, nullptr, f_attnO, QQ, DHD, TTK, BB, NHH,
        TTK, 1, (long long)NHH * QQ * TTK, (long long)QQ * TTK,
        TTK, 1, (long long)EE * TTK, (long long)DHD * TTK,
        EE, 1, (long long)QQ * EE, DHD, 0);
    launch_gemm(st, f_attnO, cw + 3 * EE * EE, cb + 3 * EE, f_t2, BB * QQ, EE, EE, 1, 1,
        EE, 1, 0, 0, EE, 1, 0, 0, EE, 1, 0, 0, 0);
    ln_kernel<<<dim3(BB * QQ), dim3(256), 0, st>>>(f_t2, f_tgt,
        ln_g + (long long)(i * 3 + 1) * EE, ln_b + (long long)(i * 3 + 1) * EE, f_tgt);

    // FFN
    launch_gemm(st, f_tgt, ffn_w1 + (long long)i * FFF * EE, ffn_b1 + (long long)i * FFF,
        f_ffh, BB * QQ, FFF, EE, 1, 1,
        EE, 1, 0, 0, EE, 1, 0, 0, FFF, 1, 0, 0, 1);
    launch_gemm(st, f_ffh, ffn_w2 + (long long)i * EE * FFF, ffn_b2 + (long long)i * EE,
        f_t2, BB * QQ, EE, FFF, 1, 1,
        FFF, 1, 0, 0, FFF, 1, 0, 0, EE, 1, 0, 0, 0);
    ln_kernel<<<dim3(BB * QQ), dim3(256), 0, st>>>(f_t2, f_tgt,
        ln_g + (long long)(i * 3 + 2) * EE, ln_b + (long long)(i * 3 + 2) * EE, f_tgt);
  }

  // ---- 8) heads ----
  float* out_class = (float*)d_out;
  float* out_mask  = (float*)d_out + (size_t)BB * QQ;
  launch_gemm(st, f_tgt, cls_w, cls_b, out_class, BB * QQ, 1, EE, 1, 1,
      EE, 1, 0, 0, EE, 1, 0, 0, 1, 1, 0, 0, 0);
  launch_gemm(st, f_tgt, me_w1, me_b1, f_m1, BB * QQ, EE, EE, 1, 1,
      EE, 1, 0, 0, EE, 1, 0, 0, EE, 1, 0, 0, 1);
  launch_gemm(st, f_m1, me_w2, me_b2, f_m2, BB * QQ, EE, EE, 1, 1,
      EE, 1, 0, 0, EE, 1, 0, 0, EE, 1, 0, 0, 1);
  launch_gemm(st, f_m2, me_w3, me_b3, f_m3, BB * QQ, EE, EE, 1, 1,
      EE, 1, 0, 0, EE, 1, 0, 0, EE, 1, 0, 0, 0);
  // outputs_mask[b,q,hw] = sum_c m3[b,q,c] * srcT[b,hw,c]
  launch_gemm(st, f_m3, f_srcT, nullptr, out_mask, QQ, HW0, EE, BB, 1,
      EE, 1, (long long)QQ * EE, 0,
      EE, 1, (long long)HW0 * EE, 0,
      HW0, 1, (long long)QQ * HW0, 0, 0);
}
